// Actor_metapop1_MDP_62878321214251
// MI455X (gfx1250) — matrix operand order repaired and checked by reading, not compiled, not run on hardware
//
#include <hip/hip_runtime.h>
#include <hip/hip_bf16.h>

typedef _Float16 half_t;
typedef __attribute__((ext_vector_type(16))) _Float16 v16h;
typedef __attribute__((ext_vector_type(8)))  float    v8f;

#define TILE_M 128
#define TILE_N 64
#define TILE_K 32
#define LDSW   20            // dwords per LDS row (16 data + 4 pad) = 80 B, 16B-aligned

// ---------------------------------------------------------------------------
// Elementwise convert: state (B x 1025 f32) -> state16 (B x 1024 f16)
// ---------------------------------------------------------------------------
__global__ __launch_bounds__(256)
void cvt_state_kernel(const float* __restrict__ s, half_t* __restrict__ d, int rows)
{
    const size_t base = ((size_t)blockIdx.x * 256 + threadIdx.x) * 8;
    if (base >= (size_t)rows * 1024) return;
    const size_t row = base >> 10, col = base & 1023;
    const float* sp = s + row * 1025 + col;
    half_t* dp = d + base;
    #pragma unroll
    for (int j = 0; j < 8; ++j) dp[j] = (half_t)sp[j];
}

// ---------------------------------------------------------------------------
// Transpose + convert: W (K x N f32, row-major) -> Wt (Npad x K f16, row-major)
// rows n >= N are zero-filled so GEMM staging never goes out of range.
// ---------------------------------------------------------------------------
__global__ __launch_bounds__(256)
void transpose_cvt_kernel(const float* __restrict__ W, half_t* __restrict__ Wt,
                          int K, int N, int Npad)
{
    __shared__ float tile[32][33];
    const int kb = blockIdx.x * 32, nb = blockIdx.y * 32;
    const int tx = threadIdx.x, ty = threadIdx.y;          // (32, 8)
    #pragma unroll
    for (int i = 0; i < 32; i += 8) {
        const int k = kb + ty + i, n = nb + tx;
        tile[ty + i][tx] = (k < K && n < N) ? W[(size_t)k * N + n] : 0.0f;
    }
    __syncthreads();
    #pragma unroll
    for (int i = 0; i < 32; i += 8) {
        const int n = nb + ty + i, k = kb + tx;
        if (n < Npad && k < K) Wt[(size_t)n * K + k] = (half_t)tile[tx][ty + i];
    }
}

// ---------------------------------------------------------------------------
// Async global->LDS B128 copy (CDNA5 path, tracked by ASYNCcnt)
// ---------------------------------------------------------------------------
__device__ __forceinline__ void async_copy16(unsigned ldsOff, const void* gaddr)
{
    asm volatile("global_load_async_to_lds_b128 %0, %1, off"
                 :: "v"(ldsOff), "v"(gaddr) : "memory");
}
__device__ __forceinline__ void async_wait0()
{
    asm volatile("s_wait_asynccnt 0x0" ::: "memory");
}

// ---------------------------------------------------------------------------
// Tiled WMMA GEMM:  C = act(A @ Bt^T + bias [+ rank1 correction])
//   A : (M x Kmain) f16 row-major, leading dim lda (elements)
//   Bt: (Npad x Kmain) f16 row-major (i.e. B transposed), leading dim Kmain
//   Block tile 128x64 (4 waves); wave computes 32x64 via 2 A-frags x 4 B-frags
//   = 8 x v_wmma_f32_16x16x32_f16 per 32-K step. Tiles are staged with
//   double-buffered async global->LDS copies so copy latency overlaps compute.
// ---------------------------------------------------------------------------
template <bool RELU, bool OUT_F16, bool KEXTRA>
__global__ __launch_bounds__(128)
void gemm_wmma_kernel(const half_t* __restrict__ A, const half_t* __restrict__ Bt,
                      const float* __restrict__ bias, void* __restrict__ Cout,
                      int N, int Kmain, int lda, int ldc,
                      const float* __restrict__ Aext, int extStride, // col 1024 of A (fp32)
                      const float* __restrict__ Bext)                // row 1024 of B (fp32)
{
    __shared__ unsigned As[2][TILE_M * LDSW];   // 2 x (128 rows x 32 f16, 80 B rows)
    __shared__ unsigned Bs[2][TILE_N * LDSW];   // 2 x ( 64 rows x 32 f16)

    const int tid  = threadIdx.x;
    const int lane = tid & 31;
    const int wave = tid >> 5;
    const int g    = lane >> 4;              // lane group (0/1)
    const int ml   = lane & 15;
    const int mBlock = blockIdx.y * TILE_M;
    const int nBlock = blockIdx.x * TILE_N;

    // Stage one 32-K tile into LDS buffer `buf` (6 async B128 copies/thread).
    auto stage = [&](int buf, int k0) {
        const unsigned asBase = (unsigned)(uintptr_t)&As[buf][0];
        const unsigned bsBase = (unsigned)(uintptr_t)&Bs[buf][0];
        #pragma unroll
        for (int i = 0; i < 4; ++i) {           // A: 128 rows x 64B = 512 chunks
            const int ci = tid + 128 * i;
            const int r = ci >> 2, c = ci & 3;
            const char* gp = (const char*)(A + (size_t)(mBlock + r) * (size_t)lda + k0) + c * 16;
            async_copy16(asBase + r * 80 + c * 16, gp);
        }
        #pragma unroll
        for (int i = 0; i < 2; ++i) {           // B: 64 n-rows x 64B = 256 chunks
            const int ci = tid + 128 * i;
            const int r = ci >> 2, c = ci & 3;
            const char* gp = (const char*)(Bt + (size_t)(nBlock + r) * (size_t)Kmain + k0) + c * 16;
            async_copy16(bsBase + r * 80 + c * 16, gp);
        }
    };

    v8f acc[2][4];
    #pragma unroll
    for (int s = 0; s < 2; ++s)
        #pragma unroll
        for (int i = 0; i < 4; ++i) acc[s][i] = (v8f){};

    // Prologue: fill buffer 0.
    stage(0, 0);
    async_wait0();
    __syncthreads();

    int buf = 0;
    for (int k0 = 0; k0 < Kmain; k0 += TILE_K, buf ^= 1) {
        const bool hasNext = (k0 + TILE_K < Kmain);
        // Kick off next tile into the other buffer (no readers of it remain).
        if (hasNext) stage(buf ^ 1, k0 + TILE_K);

        // ---- 2 A fragments (16x32 each), contiguous K per lane half ----
        union { v16h v; unsigned u[8]; } af[2];
        #pragma unroll
        for (int s = 0; s < 2; ++s) {
            const int arow = wave * 32 + s * 16 + ml;
            #pragma unroll
            for (int v = 0; v < 8; ++v) {
                const int p = v + (g << 3); // K-pair index
                af[s].u[v] = As[buf][arow * LDSW + p];
            }
        }
        // ---- 4 B fragments (32x16) x 2 subtiles -> 8 WMMAs ----
        #pragma unroll
        for (int nt = 0; nt < 4; ++nt) {
            union { v16h v; unsigned u[8]; } bf;
            const int brow = nt * 16 + ml;
            #pragma unroll
            for (int v = 0; v < 8; ++v)
                bf.u[v] = Bs[buf][brow * LDSW + v + (g << 3)];
            #pragma unroll
            for (int s = 0; s < 2; ++s)
                acc[s][nt] = __builtin_amdgcn_wmma_f32_16x16x32_f16(
                    false, af[s].v, false, bf.v, (short)0, acc[s][nt], false, false);
        }

        if (hasNext) {
            async_wait0();      // only next-tile copies are outstanding
            __syncthreads();    // next tile visible; everyone done reading current
        }
    }

    // ---- epilogue: bias, optional K=1025 rank-1 correction, ReLU, store ----
    float aext[2][8];
    if constexpr (KEXTRA) {
        #pragma unroll
        for (int s = 0; s < 2; ++s)
            #pragma unroll
            for (int r = 0; r < 8; ++r) {
                const int row = mBlock + wave * 32 + s * 16 + r + 8 * g;
                aext[s][r] = Aext[(size_t)row * (size_t)extStride];
            }
    }
    #pragma unroll
    for (int nt = 0; nt < 4; ++nt) {
        const int col = nBlock + nt * 16 + ml;
        const bool colOk = (col < N);
        const float bv = colOk ? bias[col] : 0.0f;
        float be = 0.0f;
        if constexpr (KEXTRA) be = Bext[col];
        #pragma unroll
        for (int s = 0; s < 2; ++s) {
            #pragma unroll
            for (int r = 0; r < 8; ++r) {
                const int row = mBlock + wave * 32 + s * 16 + r + 8 * g;
                float c = acc[s][nt][r] + bv;
                if constexpr (KEXTRA) c += aext[s][r] * be;
                if constexpr (RELU)   c = fmaxf(c, 0.0f);
                if (colOk) {
                    if constexpr (OUT_F16)
                        ((half_t*)Cout)[(size_t)row * (size_t)ldc + col] = (half_t)c;
                    else
                        ((float*)Cout)[(size_t)row * (size_t)ldc + col] = c;
                }
            }
        }
    }
}

// ---------------------------------------------------------------------------
// Without-replacement logprob/entropy scan. One wave32 per batch row; the wave
// handles both halves (R then S). N=513 logits live in 17 register slots/lane
// (j = 32*s + lane), plus an LDS copy for the random x[idx] fetch.
// Identity: with fixed max m and T=sum(e), U=sum(e*x) over remaining,
//   lse = m + log T ; ent_step = lse - U/T ; logp += x[idx] - lse.
// ---------------------------------------------------------------------------
#define NHEAD 513
#define KSEQ  32

__global__ __launch_bounds__(128)
void scan_kernel(const float* __restrict__ logits,   // (Brows, 1026)
                 const int* __restrict__ idxR, const int* __restrict__ lenR,
                 const int* __restrict__ idxS, const int* __restrict__ lenS,
                 float* __restrict__ out, int Brows)
{
    __shared__ float xs[4][520];
    const int lane = threadIdx.x & 31;
    const int wave = threadIdx.x >> 5;
    const int row  = blockIdx.x * 4 + wave;
    if (row >= Brows) return;

    float totLogp = 0.0f, totEnt = 0.0f;

    for (int half = 0; half < 2; ++half) {
        const float* x    = logits + (size_t)row * 1026 + half * NHEAD;
        const int*   sidx = (half ? idxS : idxR) + row * KSEQ;
        const int    slen = (half ? lenS : lenR)[row];

        float xv[17], ev[17];
        float m = -3.4e38f;
        #pragma unroll
        for (int s = 0; s < 17; ++s) {
            const int j = s * 32 + lane;
            const float v = (j < NHEAD) ? x[j] : -3.4e38f;
            xv[s] = v;
            if (j < NHEAD) xs[wave][j] = v;
            m = fmaxf(m, v);
        }
        #pragma unroll
        for (int o = 16; o > 0; o >>= 1) m = fmaxf(m, __shfl_xor(m, o, 32));
        #pragma unroll
        for (int s = 0; s < 17; ++s) {
            const int j = s * 32 + lane;
            ev[s] = (j < NHEAD) ? expf(xv[s] - m) : 0.0f;
        }

        unsigned removed = 0;
        for (int t = 0; t < KSEQ; ++t) {
            if (t >= slen) break;                 // wave-uniform
            float tl = 0.0f, ul = 0.0f;
            #pragma unroll
            for (int s = 0; s < 17; ++s) {
                const float e = ((removed >> s) & 1u) ? 0.0f : ev[s];
                tl += e;
                ul += e * xv[s];
            }
            #pragma unroll
            for (int o = 16; o > 0; o >>= 1) {
                tl += __shfl_xor(tl, o, 32);
                ul += __shfl_xor(ul, o, 32);
            }
            const float lse = m + logf(tl);
            totEnt += lse - ul / tl;
            const int idx = sidx[t];              // uniform scalar load
            if (idx >= 0) {
                totLogp += xs[wave][idx] - lse;
                if ((idx & 31) == lane) removed |= 1u << (idx >> 5);
            }
        }
    }

    if (lane == 0) {
        out[row]         = totLogp;
        out[Brows + row] = totEnt;
    }
}

// ---------------------------------------------------------------------------
extern "C" void kernel_launch(void* const* d_in, const int* in_sizes, int n_in,
                              void* d_out, int out_size, void* d_ws, size_t ws_size,
                              hipStream_t stream) {
    (void)in_sizes; (void)n_in; (void)out_size; (void)ws_size;
    const int Brows = 8192, DIN = 1025, H = 1024, NACT = 1026, NPAD2 = 1088;

    const float* state = (const float*)d_in[0];
    const float* W0    = (const float*)d_in[1];
    const float* b0    = (const float*)d_in[2];
    const float* W1    = (const float*)d_in[3];
    const float* b1    = (const float*)d_in[4];
    const float* W2    = (const float*)d_in[5];
    const float* b2    = (const float*)d_in[6];
    const int* idxR    = (const int*)d_in[7];
    const int* lenR    = (const int*)d_in[8];
    const int* idxS    = (const int*)d_in[9];
    const int* lenS    = (const int*)d_in[10];
    float* out = (float*)d_out;

    // workspace layout (f16 unless noted), 256B-aligned segments
    char* ws = (char*)d_ws;
    size_t off = 0;
    half_t* s16 = (half_t*)(ws + off); off += (size_t)Brows * H * 2;        // 16 MiB
    half_t* Wt0 = (half_t*)(ws + off); off += (size_t)H * H * 2;            //  2 MiB
    half_t* Wt1 = (half_t*)(ws + off); off += (size_t)H * H * 2;            //  2 MiB
    half_t* Wt2 = (half_t*)(ws + off); off += (size_t)NPAD2 * H * 2;        // ~2 MiB
    half_t* h0  = (half_t*)(ws + off); off += (size_t)Brows * H * 2;        // 16 MiB
    half_t* h1  = (half_t*)(ws + off); off += (size_t)Brows * H * 2;        // 16 MiB
    float* logits = (float*)(ws + off);                                     // 32 MiB

    // ---- stage 0: one-time f16 conversion (+ weight transpose) ----
    cvt_state_kernel<<<dim3((Brows * H) / (256 * 8)), dim3(256), 0, stream>>>(state, s16, Brows);
    dim3 tb(32, 8);
    transpose_cvt_kernel<<<dim3(H / 32, H / 32),     tb, 0, stream>>>(W0, Wt0, H, H, H);
    transpose_cvt_kernel<<<dim3(H / 32, H / 32),     tb, 0, stream>>>(W1, Wt1, H, H, H);
    transpose_cvt_kernel<<<dim3(H / 32, NPAD2 / 32), tb, 0, stream>>>(W2, Wt2, H, NACT, NPAD2);

    dim3 blk(128);
    const int MT = Brows / TILE_M;
    // GEMM0: relu(state @ W0 + b0); K = 1024 WMMA + fp32 rank-1 term for k=1024
    gemm_wmma_kernel<true, true, true>
        <<<dim3(H / TILE_N, MT), blk, 0, stream>>>(
            s16, Wt0, b0, (void*)h0, H, H, H, H,
            state + 1024, DIN, W0 + (size_t)1024 * H);
    // GEMM1: relu(h0 @ W1 + b1)
    gemm_wmma_kernel<true, true, false>
        <<<dim3(H / TILE_N, MT), blk, 0, stream>>>(
            h0, Wt1, b1, (void*)h1, H, H, H, H, nullptr, 0, nullptr);
    // GEMM2: logits = h1 @ W2 + b2   (N = 1026, Bt zero-padded to 1088 rows)
    gemm_wmma_kernel<false, false, false>
        <<<dim3(NPAD2 / TILE_N, MT), blk, 0, stream>>>(
            h1, Wt2, b2, (void*)logits, NACT, H, H, NACT, nullptr, 0, nullptr);
    // Scan: logp/entropy without replacement for both halves
    scan_kernel<<<dim3(Brows / 4), blk, 0, stream>>>(
        logits, idxR, lenR, idxS, lenS, out, Brows);
}
